// BayesianNN_31464930411175
// MI455X (gfx1250) — matrix operand order repaired and checked by reading, not compiled, not run on hardware
//
#include <hip/hip_runtime.h>
#include <hip/hip_bf16.h>

// ---------------------------------------------------------------------------
// Types
// ---------------------------------------------------------------------------
typedef __attribute__((ext_vector_type(16))) _Float16 v16h;
typedef __attribute__((ext_vector_type(8)))  _Float16 v8h;
typedef __attribute__((ext_vector_type(8)))  float    v8f;

#define S_SAMPLES 32
#define BATCH     2048

static constexpr int LDS_PITCH = 72;   // 64 halves K-step + 8 pad -> conflict-free 16B reads
static constexpr int K_STEP    = 64;   // two WMMA K-slices per LDS stage

// ---------------------------------------------------------------------------
// Fast hardware tanh (V_TANH_F32, gfx1250 TRANS op)
// ---------------------------------------------------------------------------
__device__ __forceinline__ float fast_tanh(float x) {
#if __has_builtin(__builtin_amdgcn_tanhf)
    return __builtin_amdgcn_tanhf(x);
#elif __has_builtin(__builtin_amdgcn_tanh_f32)
    return __builtin_amdgcn_tanh_f32(x);
#else
    float r;
    // TRANS-op hazard: 1 independent op / V_NOP before result use (ISA 7.4)
    asm volatile("v_tanh_f32 %0, %1\n\tv_nop" : "=v"(r) : "v"(x));
    return r;
#endif
}

// ---------------------------------------------------------------------------
// Prep kernel 1: fp32 -> f16 conversion (x, broadcast over samples)
// ---------------------------------------------------------------------------
__global__ void cvt_f32_f16_kernel(const float* __restrict__ in,
                                   _Float16* __restrict__ out, int n4) {
    int i = blockIdx.x * blockDim.x + threadIdx.x;
    if (i < n4) {
        float4 v = reinterpret_cast<const float4*>(in)[i];
        _Float16 h[4] = {(_Float16)v.x, (_Float16)v.y, (_Float16)v.z, (_Float16)v.w};
        *reinterpret_cast<uint2*>(out + 4 * (size_t)i) = *reinterpret_cast<uint2*>(h);
    }
}

// ---------------------------------------------------------------------------
// Prep kernel 2: materialize per-sample weights, TRANSPOSED to [s][n][k], f16
//   W[s,k,n] = z_w[s,k,n] * exp(w_ls[k,n]) + w_m[k,n]
// ---------------------------------------------------------------------------
__global__ void make_wt_kernel(const float* __restrict__ zw,
                               const float* __restrict__ wls,
                               const float* __restrict__ wm,
                               _Float16* __restrict__ wt, int K, int N) {
    __shared__ _Float16 tile[32][33];
    const int s  = blockIdx.z;
    const int n0 = blockIdx.x * 32;
    const int k0 = blockIdx.y * 32;
    const int tx = threadIdx.x, ty = threadIdx.y;   // block (32,8)
#pragma unroll
    for (int j = 0; j < 4; ++j) {
        int k = k0 + ty + 8 * j;
        int n = n0 + tx;
        size_t o  = ((size_t)s * K + k) * N + n;
        size_t ow = (size_t)k * N + n;
        float v = zw[o] * __expf(wls[ow]) + wm[ow];
        tile[ty + 8 * j][tx] = (_Float16)v;
    }
    __syncthreads();
#pragma unroll
    for (int j = 0; j < 4; ++j) {
        int n = n0 + ty + 8 * j;
        int k = k0 + tx;
        wt[((size_t)s * N + n) * K + k] = tile[tx][ty + 8 * j];
    }
}

// ---------------------------------------------------------------------------
// Prep kernel 3: biases, fp32: b[s,n] = z_b[s,0,n]*exp(b_ls[n]) + b_m[n]
// ---------------------------------------------------------------------------
__global__ void make_bias_kernel(const float* __restrict__ zb,
                                 const float* __restrict__ bls,
                                 const float* __restrict__ bm,
                                 float* __restrict__ bias, int N, int total) {
    int i = blockIdx.x * blockDim.x + threadIdx.x;
    if (i < total) {
        int n = i % N;
        bias[i] = zb[i] * __expf(bls[n]) + bm[n];
    }
}

// ---------------------------------------------------------------------------
// WMMA fragment loads from LDS (layouts per CDNA5 ISA 7.12.2, wave32)
// ---------------------------------------------------------------------------
__device__ __forceinline__ v16h cat16(v8h lo, v8h hi) {
    return __builtin_shufflevector(lo, hi, 0, 1, 2, 3, 4, 5, 6, 7,
                                   8, 9, 10, 11, 12, 13, 14, 15);
}

// A (16x32 f16): lane L -> row (L&15); halves: K = (L>=16)*8 + [0..7], then +16
__device__ __forceinline__ v16h ld_fragA(const _Float16* sA, int rowBase, int lane, int kk) {
    int off = (rowBase + (lane & 15)) * LDS_PITCH + kk + ((lane >> 4) << 3);
    v8h lo = *reinterpret_cast<const v8h*>(sA + off);
    v8h hi = *reinterpret_cast<const v8h*>(sA + off + 16);
    return cat16(lo, hi);
}

// B (32x16 f16), stored n-major: lane L -> col (L&15); halves: K = (L>=16)*8 + [0..7], then +16
__device__ __forceinline__ v16h ld_fragB(const _Float16* sB, int nBase, int lane, int kk) {
    int off = (nBase + (lane & 15)) * LDS_PITCH + kk + ((lane >> 4) << 3);
    v8h lo = *reinterpret_cast<const v8h*>(sB + off);
    v8h hi = *reinterpret_cast<const v8h*>(sB + off + 16);
    return cat16(lo, hi);
}

// Raw LDS byte offset from a generic pointer to __shared__ (low 32 bits,
// ISA 10.2 aperture mapping).
__device__ __forceinline__ uint32_t lds_off_u32(const void* p) {
    return (uint32_t)(uintptr_t)p;
}

// Async global->LDS 16B copy (GVS mode), tracked by ASYNCcnt.
__device__ __forceinline__ void async_copy_b128(uint32_t ldsByteAddr,
                                                uint32_t gByteOff,
                                                const void* sBase) {
    asm volatile("global_load_async_to_lds_b128 %0, %1, %2"
                 :: "v"(ldsByteAddr), "v"(gByteOff), "s"(sBase)
                 : "memory");
}

__device__ __forceinline__ void wait_async0() {
    asm volatile("s_wait_asynccnt 0x0" ::: "memory");
}

// ---------------------------------------------------------------------------
// Batched GEMM:  out[s] = act( A[s] (MxK) * Wt[s]^T (KxN) + bias[s] )
//   A: f16 row-major (pitch K), optionally broadcast over s (aStride==0)
//   Wt: f16, [s][n][k] (pitch K)
//   8 waves (256 thr), block tile M_BLK x N_BLK, double-buffered async LDS
// ---------------------------------------------------------------------------
template <int M_BLK, int N_BLK, int WAVES_M, int WAVES_N, bool ACT, bool OUT_F32>
__global__ __launch_bounds__(256) void gemm_wmma_kernel(
        const _Float16* __restrict__ A, const _Float16* __restrict__ Wt,
        const float* __restrict__ bias, void* __restrict__ out,
        int M, int N, int K, size_t aStride) {
    constexpr int WM = M_BLK / WAVES_M;       // wave tile M
    constexpr int WN = N_BLK / WAVES_N;       // wave tile N
    constexpr int FM = WM / 16;               // 16x16 frags per wave (M)
    constexpr int FN = WN / 16;               // 16x16 frags per wave (N)

    __shared__ alignas(16) _Float16 sA[2][M_BLK * LDS_PITCH];
    __shared__ alignas(16) _Float16 sB[2][N_BLK * LDS_PITCH];

    const int s     = blockIdx.z;
    const int mBase = blockIdx.y * M_BLK;
    const int nBase = blockIdx.x * N_BLK;
    const int tid   = threadIdx.x;
    const int lane  = tid & 31;
    const int wave  = tid >> 5;
    const int wm    = wave / WAVES_N;
    const int wn    = wave % WAVES_N;

    const _Float16* __restrict__ aS = A  + (size_t)s * aStride;
    const _Float16* __restrict__ bS = Wt + (size_t)s * (size_t)N * (size_t)K;
    const float*    __restrict__ biasS = bias + (size_t)s * N;

    const uint32_t ldsA0 = lds_off_u32(&sA[0][0]);
    const uint32_t ldsA1 = lds_off_u32(&sA[1][0]);
    const uint32_t ldsB0 = lds_off_u32(&sB[0][0]);
    const uint32_t ldsB1 = lds_off_u32(&sB[1][0]);

    // Issue async global->LDS copies for one K_STEP-wide tile pair.
    auto issue_tiles = [&](int buf, int k0) {
        const uint32_t aBase = buf ? ldsA1 : ldsA0;
        const uint32_t bBase = buf ? ldsB1 : ldsB0;
        for (int i = tid; i < M_BLK * 8; i += 256) {       // A: M_BLK x 64 halves
            int r = i >> 3, c = i & 7;
            uint32_t lds = aBase + (uint32_t)((r * LDS_PITCH + c * 8) * 2);
            uint32_t gof = (uint32_t)((((size_t)(mBase + r)) * K + k0 + c * 8) * 2);
            async_copy_b128(lds, gof, aS);
        }
        for (int i = tid; i < N_BLK * 8; i += 256) {       // B: N_BLK x 64 halves
            int r = i >> 3, c = i & 7;
            uint32_t lds = bBase + (uint32_t)((r * LDS_PITCH + c * 8) * 2);
            uint32_t gof = (uint32_t)((((size_t)(nBase + r)) * K + k0 + c * 8) * 2);
            async_copy_b128(lds, gof, bS);
        }
    };

    v8f acc[FM][FN];
#pragma unroll
    for (int i = 0; i < FM; ++i)
#pragma unroll
        for (int j = 0; j < FN; ++j) acc[i][j] = (v8f)0.0f;

    issue_tiles(0, 0);
    int buf = 0;
    for (int k0 = 0; k0 < K; k0 += K_STEP) {
        wait_async0();        // my async writes for `buf` are in LDS
        __syncthreads();      // everyone's are; everyone done reading buf^1
        if (k0 + K_STEP < K) issue_tiles(buf ^ 1, k0 + K_STEP);

        const _Float16* cA = &sA[buf][0];
        const _Float16* cB = &sB[buf][0];
#pragma unroll
        for (int kk = 0; kk < K_STEP; kk += 32) {
            v16h af[FM], bf[FN];
#pragma unroll
            for (int i = 0; i < FM; ++i)
                af[i] = ld_fragA(cA, wm * WM + i * 16, lane, kk);
#pragma unroll
            for (int j = 0; j < FN; ++j)
                bf[j] = ld_fragB(cB, wn * WN + j * 16, lane, kk);
#pragma unroll
            for (int i = 0; i < FM; ++i)
#pragma unroll
                for (int j = 0; j < FN; ++j)
                    acc[i][j] = __builtin_amdgcn_wmma_f32_16x16x32_f16(
                        false, af[i], false, bf[j], (short)0, acc[i][j],
                        false, false);
        }
        buf ^= 1;
    }

    // Epilogue: bias + optional tanh; C layout: VGPR r -> row r + 8*(lane>=16)
#pragma unroll
    for (int i = 0; i < FM; ++i) {
#pragma unroll
        for (int j = 0; j < FN; ++j) {
            int row0 = mBase + wm * WM + i * 16 + ((lane >> 4) << 3);
            int col  = nBase + wn * WN + j * 16 + (lane & 15);
            float bv = biasS[col];
            v8f c = acc[i][j];
#pragma unroll
            for (int r = 0; r < 8; ++r) {
                float v = c[r] + bv;
                if (ACT) v = fast_tanh(v);
                size_t o = ((size_t)s * M + row0 + r) * N + col;
                if (OUT_F32)
                    reinterpret_cast<float*>(out)[o] = v;
                else
                    reinterpret_cast<_Float16*>(out)[o] = (_Float16)v;
            }
        }
    }
}

// ---------------------------------------------------------------------------
// Host launcher
// ---------------------------------------------------------------------------
static inline char* carve(char*& p, size_t bytes) {
    char* r = p;
    p += (bytes + 255) & ~size_t(255);
    return r;
}

extern "C" void kernel_launch(void* const* d_in, const int* in_sizes, int n_in,
                              void* d_out, int out_size, void* d_ws, size_t ws_size,
                              hipStream_t stream) {
    (void)in_sizes; (void)n_in; (void)out_size; (void)ws_size;

    const float* x = (const float*)d_in[0];
    // Per layer: w_mean, w_log_std, b_mean, b_log_std, z_w, z_b
    const float* wmean[3] = {(const float*)d_in[1], (const float*)d_in[7],  (const float*)d_in[13]};
    const float* wls[3]   = {(const float*)d_in[2], (const float*)d_in[8],  (const float*)d_in[14]};
    const float* bmean[3] = {(const float*)d_in[3], (const float*)d_in[9],  (const float*)d_in[15]};
    const float* bls[3]   = {(const float*)d_in[4], (const float*)d_in[10], (const float*)d_in[16]};
    const float* zw[3]    = {(const float*)d_in[5], (const float*)d_in[11], (const float*)d_in[17]};
    const float* zb[3]    = {(const float*)d_in[6], (const float*)d_in[12], (const float*)d_in[18]};

    const int Kd[3] = {256, 512, 512};
    const int Nd[3] = {512, 512, 64};

    char* p = (char*)d_ws;
    _Float16* xh  = (_Float16*)carve(p, (size_t)BATCH * 256 * 2);
    _Float16* wt[3];
    float*    bb[3];
    for (int l = 0; l < 3; ++l)
        wt[l] = (_Float16*)carve(p, (size_t)S_SAMPLES * Kd[l] * Nd[l] * 2);
    for (int l = 0; l < 3; ++l)
        bb[l] = (float*)carve(p, (size_t)S_SAMPLES * Nd[l] * 4);
    _Float16* h1 = (_Float16*)carve(p, (size_t)S_SAMPLES * BATCH * 512 * 2);
    _Float16* h2 = (_Float16*)carve(p, (size_t)S_SAMPLES * BATCH * 512 * 2);

    // --- Prep ---
    {
        int n4 = BATCH * 256 / 4;
        cvt_f32_f16_kernel<<<(n4 + 255) / 256, 256, 0, stream>>>(x, xh, n4);
    }
    for (int l = 0; l < 3; ++l) {
        dim3 g(Nd[l] / 32, Kd[l] / 32, S_SAMPLES);
        make_wt_kernel<<<g, dim3(32, 8), 0, stream>>>(zw[l], wls[l], wmean[l],
                                                      wt[l], Kd[l], Nd[l]);
        int tot = S_SAMPLES * Nd[l];
        make_bias_kernel<<<(tot + 255) / 256, 256, 0, stream>>>(zb[l], bls[l],
                                                                bmean[l], bb[l],
                                                                Nd[l], tot);
    }

    // --- Layer 0: (2048x256, broadcast A) x (256x512) -> tanh -> h1 (f16) ---
    {
        dim3 g(Nd[0] / 128, BATCH / 128, S_SAMPLES);
        gemm_wmma_kernel<128, 128, 2, 4, true, false>
            <<<g, 256, 0, stream>>>(xh, wt[0], bb[0], h1,
                                    BATCH, Nd[0], Kd[0], /*aStride=*/0);
    }
    // --- Layer 1: (2048x512) x (512x512) -> tanh -> h2 (f16) ---
    {
        dim3 g(Nd[1] / 128, BATCH / 128, S_SAMPLES);
        gemm_wmma_kernel<128, 128, 2, 4, true, false>
            <<<g, 256, 0, stream>>>(h1, wt[1], bb[1], h2,
                                    BATCH, Nd[1], Kd[1],
                                    (size_t)BATCH * Kd[1]);
    }
    // --- Layer 2: (2048x512) x (512x64) -> fp32 d_out ---
    {
        dim3 g(Nd[2] / 64, BATCH / 128, S_SAMPLES);
        gemm_wmma_kernel<128, 64, 4, 2, false, true>
            <<<g, 256, 0, stream>>>(h2, wt[2], bb[2], d_out,
                                    BATCH, Nd[2], Kd[2],
                                    (size_t)BATCH * Kd[2]);
    }
}
